// AdaptiveGridAttention_43430709297884
// MI455X (gfx1250) — matrix operand order repaired and checked by reading, not compiled, not run on hardware
//
#include <hip/hip_runtime.h>
#include <hip/hip_bf16.h>

typedef _Float16 f16;
typedef __attribute__((ext_vector_type(16))) _Float16 v16h;
typedef __attribute__((ext_vector_type(8)))  _Float16 v8h;
typedef __attribute__((ext_vector_type(8)))  float    v8f;
typedef int vsi4 __attribute__((vector_size(16)));   // matches builtin param type

union U16 { v16h v; v8h h[2]; };

__device__ __forceinline__ v8f wmma_f16(v16h a, v16h b, v8f c) {
    // D = A(16x32 f16) * B(32x16 f16) + C(16x16 f32)
    return __builtin_amdgcn_wmma_f32_16x16x32_f16(
        /*neg_a=*/false, a, /*neg_b=*/false, b,
        /*c_mod=*/(short)0, c, /*reuse_a=*/false, /*reuse_b=*/false);
}

// ---- CDNA5 async global->LDS copy (16 bytes per lane), ASYNCcnt-tracked ----
__device__ __forceinline__ void async_copy16(const f16* g, f16* l) {
#if defined(__AMDGCN__)
#if __has_builtin(__builtin_amdgcn_global_load_async_to_lds_b128)
    __builtin_amdgcn_global_load_async_to_lds_b128(
        (__attribute__((address_space(1))) vsi4*)g,
        (__attribute__((address_space(3))) vsi4*)l,
        /*imm offset*/0, /*cpol*/0);
#else
    unsigned loff = (unsigned)(unsigned long)
        (__attribute__((address_space(3))) void*)l;
    asm volatile("global_load_async_to_lds_b128 %0, %1, off"
                 :: "v"(loff), "v"(g) : "memory");
#endif
#else
    *(v8h*)l = *(const v8h*)g;   // host-pass stub (never executed)
#endif
}

__device__ __forceinline__ void wait_async0() {
#if defined(__AMDGCN__)
#if __has_builtin(__builtin_amdgcn_s_wait_asynccnt)
    __builtin_amdgcn_s_wait_asynccnt(0);
#else
    asm volatile("s_wait_asynccnt 0x0" ::: "memory");
#endif
#endif
}

// ---------------------------------------------------------------------------
// Problem constants: x (2,128,126,126) f32; C=128, NH=8, BS=8, g=16
// L = 2*16*16 = 512 windows (the MHA "sequence"), N = 64 (window positions,
// the MHA "batch"), dh = 16.  Rows R = L*N = 32768.
// ---------------------------------------------------------------------------

// Kernel 0: convert packed weights to f16 (K-contiguous rows already).
__global__ __launch_bounds__(256) void prep_weights(
    const float* __restrict__ w_in, const float* __restrict__ w_out,
    f16* __restrict__ Wi, f16* __restrict__ Wo) {
    int tid = blockIdx.x * 256 + threadIdx.x;
    if (tid < 49152) Wi[tid] = (f16)w_in[tid];       // (384,128)
    if (tid < 16384) Wo[tid] = (f16)w_out[tid];      // (128,128)
}

// Kernel 1: window + zero-pad + pre-scale (xw * dh^-0.5 = 0.25) into
// Xw[row=(l*64+n)][c] f16, row-major over c (so GEMM-A loads are contiguous).
__global__ __launch_bounds__(256) void window_kernel(
    const float* __restrict__ x, f16* __restrict__ Xw) {
    int tid = blockIdx.x * 256 + threadIdx.x;        // 0 .. 32768*128-1
    int c   = tid & 127;
    int row = tid >> 7;
    int l = row >> 6, n = row & 63;
    int b = l >> 8, rem = l & 255;
    int gy = rem >> 4, gx = rem & 15;
    int by = n >> 3,  bx = n & 7;
    int y = gy * 8 + by, xx = gx * 8 + bx;
    float v = 0.f;
    if (y < 126 && xx < 126)
        v = x[(((size_t)b * 128 + c) * 126 + y) * 126 + xx] * 0.25f;
    Xw[tid] = (f16)v;
}

// Kernel 2: qkv projection GEMM (32768 x 384 x K=128) via WMMA f16.
// One 16x16 output tile per wave; scatter into attention layouts:
//   Qp[n][h][l][0..31] (d>=16 zero-padded, extra 0.25 q-scale + bias)
//   Kp[n][h][l][0..31] (zero-padded)
//   Vt[n][h][d][l]     (V transposed: dh-major rows, l contiguous)
__global__ __launch_bounds__(256) void qkv_gemm(
    const f16* __restrict__ Xw, const f16* __restrict__ Wi,
    const float* __restrict__ b_in,
    f16* __restrict__ Qp, f16* __restrict__ Kp, f16* __restrict__ Vt) {
    const int lane = threadIdx.x & 31;
    const int wave = threadIdx.x >> 5;
    const int tileM = blockIdx.x * 8 + wave;   // 0..2047
    const int tileN = blockIdx.y;              // 0..23
    const int half = lane >> 4;
    const int l16  = lane & 15;
    const int rowA = tileM * 16 + l16;
    const int j    = tileN * 16 + l16;         // global output column
    v8f acc = {};
    #pragma unroll
    for (int kt = 0; kt < 4; ++kt) {
        const int kb = kt * 32;
        U16 a, b;
        const f16* ap = Xw + (size_t)rowA * 128 + kb + half * 8;
        a.h[0] = *(const v8h*)(ap);
        a.h[1] = *(const v8h*)(ap + 16);
        const f16* bp = Wi + (size_t)j * 128 + kb + half * 8;
        b.h[0] = *(const v8h*)(bp);
        b.h[1] = *(const v8h*)(bp + 16);
        acc = wmma_f16(a.v, b.v, acc);
    }
    const float bias = b_in[j];
    #pragma unroll
    for (int r = 0; r < 8; ++r) {
        int row = tileM * 16 + r + half * 8;
        int l = row >> 6, n = row & 63;
        float v = acc[r] + bias;
        if (j < 128) {                               // Q (+ MHA's own dh^-0.5)
            int hh = j >> 4, d = j & 15;
            size_t base = ((size_t)(n * 8 + hh) * 512 + l) << 5;
            Qp[base + d]      = (f16)(v * 0.25f);
            Qp[base + 16 + d] = (f16)0.f;            // pad dh 16->32
        } else if (j < 256) {                        // K
            int jj = j - 128; int hh = jj >> 4, d = jj & 15;
            size_t base = ((size_t)(n * 8 + hh) * 512 + l) << 5;
            Kp[base + d]      = (f16)v;
            Kp[base + 16 + d] = (f16)0.f;
        } else {                                     // V (transposed)
            int jj = j - 256; int hh = jj >> 4, d = jj & 15;
            Vt[((size_t)(n * 8 + hh) * 16 + d) * 512 + l] = (f16)v;
        }
    }
}

// Kernel 3: per-(n,h) attention.  S^T = K Q^T keeps each query in one lane
// (per-lane softmax stats, two shfl_xor(16) combines).  O^T = V^T P with the
// P operand built in-register (one half-wave swap).  Q/K/V staged in LDS via
// CDNA5 async global->LDS loads (ASYNCcnt), then consumed through ds reads.
__global__ __launch_bounds__(256) void attn_kernel(
    const f16* __restrict__ Qp, const f16* __restrict__ Kp,
    const f16* __restrict__ Vt, f16* __restrict__ Of) {
    extern __shared__ __align__(16) f16 smem[];
    f16* Qs = smem;               // 512*32 halves  (32 KB)
    f16* Ks = smem + 16384;       // 512*32 halves  (32 KB)
    f16* Vs = smem + 32768;       // 16*512 halves  (16 KB)
    const int nh = blockIdx.x;    // n*8 + h
    const int n = nh >> 3, h = nh & 7;
    const int tid = threadIdx.x;
    {   // async staging: all three source blocks are contiguous per (n,h)
        const f16* s0 = Qp + (size_t)nh * 16384;
        const f16* s1 = Kp + (size_t)nh * 16384;
        const f16* s2 = Vt + (size_t)nh * 8192;
        for (int i = tid; i < 2048; i += 256) async_copy16(s0 + i * 8, Qs + i * 8);
        for (int i = tid; i < 2048; i += 256) async_copy16(s1 + i * 8, Ks + i * 8);
        for (int i = tid; i < 1024; i += 256) async_copy16(s2 + i * 8, Vs + i * 8);
    }
    wait_async0();                // ASYNCcnt == 0: LDS writes landed
    __syncthreads();
    const int lane = tid & 31;
    const int wave = tid >> 5;
    const int half = lane >> 4;
    const int l16  = lane & 15;
    for (int ch = wave; ch < 32; ch += 8) {          // 16-query chunks
        const int q0 = ch * 16;
        U16 bq;                                      // B = Q^T (32dh x 16q)
        const f16* qp = Qs + (size_t)(q0 + l16) * 32 + half * 8;
        bq.h[0] = *(const v8h*)(qp);
        bq.h[1] = *(const v8h*)(qp + 16);
        // ---- pass A: per-lane running max over this lane's half of keys
        float m = -1e30f;
        for (int t = 0; t < 32; ++t) {
            U16 ak;                                  // A = K tile (16k x 32dh)
            const f16* kp = Ks + (size_t)(t * 16 + l16) * 32 + half * 8;
            ak.h[0] = *(const v8h*)(kp);
            ak.h[1] = *(const v8h*)(kp + 16);
            v8f s = {};
            s = wmma_f16(ak.v, bq.v, s);
            #pragma unroll
            for (int r = 0; r < 8; ++r) m = fmaxf(m, s[r]);
        }
        m = fmaxf(m, __shfl_xor(m, 16, 32));         // full row max per query
        // ---- pass B: exp, sum, and P@V accumulation
        float lsum = 0.f;
        v8f ot = {};
        for (int t = 0; t < 32; t += 2) {
            v8f s0 = {}, s1 = {};
            {
                U16 ak;
                const f16* kp = Ks + (size_t)(t * 16 + l16) * 32 + half * 8;
                ak.h[0] = *(const v8h*)(kp); ak.h[1] = *(const v8h*)(kp + 16);
                s0 = wmma_f16(ak.v, bq.v, s0);
                kp = Ks + (size_t)((t + 1) * 16 + l16) * 32 + half * 8;
                ak.h[0] = *(const v8h*)(kp); ak.h[1] = *(const v8h*)(kp + 16);
                s1 = wmma_f16(ak.v, bq.v, s1);
            }
            unsigned int a4[4], b4[4];
            #pragma unroll
            for (int r = 0; r < 4; ++r) {
                float p0 = __expf(s0[2 * r] - m), p1 = __expf(s0[2 * r + 1] - m);
                lsum += p0 + p1;
                union { f16 x[2]; unsigned int u; } pk;
                pk.x[0] = (f16)p0; pk.x[1] = (f16)p1;
                a4[r] = pk.u;
                p0 = __expf(s1[2 * r] - m); p1 = __expf(s1[2 * r + 1] - m);
                lsum += p0 + p1;
                pk.x[0] = (f16)p0; pk.x[1] = (f16)p1;
                b4[r] = pk.u;
            }
            // B operand for PV: (32 keys x 16 queries); half-wave swap only
            union { v16h v; unsigned int u[8]; } bp;
            #pragma unroll
            for (int r = 0; r < 4; ++r) {
                unsigned int oa = __shfl_xor(a4[r], 16, 32);
                unsigned int ob = __shfl_xor(b4[r], 16, 32);
                bp.u[r]     = half ? ob    : a4[r];
                bp.u[4 + r] = half ? b4[r] : oa;
            }
            U16 av;                                  // A = V^T tile (16dh x 32k)
            const f16* vp = Vs + (size_t)l16 * 512;
            av.h[0] = *(const v8h*)(vp + t * 16 + half * 16);
            av.h[1] = *(const v8h*)(vp + t * 16 + half * 16 + 8);
            ot = wmma_f16(av.v, bp.v, ot);
        }
        lsum += __shfl_xor(lsum, 16, 32);            // full row sum per query
        const float inv = 1.f / lsum;
        // O^T C-layout: lane query = l16, rows = dh r + 8*half -> contiguous
        const int q = q0 + l16;
        v8h o8;
        #pragma unroll
        for (int r = 0; r < 8; ++r) o8[r] = (f16)(ot[r] * inv);
        *(v8h*)(Of + ((size_t)q * 64 + n) * 128 + h * 16 + half * 8) = o8;
    }
}

// Kernel 4: output projection GEMM (32768 x 128 x K=128) + bias + un-window
// scatter to (2,128,126,126) f32; padded rows are dropped.
__global__ __launch_bounds__(256) void out_gemm(
    const f16* __restrict__ Of, const f16* __restrict__ Wo,
    const float* __restrict__ b_out, float* __restrict__ out) {
    const int lane = threadIdx.x & 31;
    const int wave = threadIdx.x >> 5;
    const int tileM = blockIdx.x * 8 + wave;   // 0..2047
    const int tileN = blockIdx.y;              // 0..7
    const int half = lane >> 4;
    const int l16  = lane & 15;
    const int rowA = tileM * 16 + l16;
    const int j    = tileN * 16 + l16;
    v8f acc = {};
    #pragma unroll
    for (int kt = 0; kt < 4; ++kt) {
        const int kb = kt * 32;
        U16 a, b;
        const f16* ap = Of + (size_t)rowA * 128 + kb + half * 8;
        a.h[0] = *(const v8h*)(ap);
        a.h[1] = *(const v8h*)(ap + 16);
        const f16* bp = Wo + (size_t)j * 128 + kb + half * 8;
        b.h[0] = *(const v8h*)(bp);
        b.h[1] = *(const v8h*)(bp + 16);
        acc = wmma_f16(a.v, b.v, acc);
    }
    const float bias = b_out[j];
    #pragma unroll
    for (int r = 0; r < 8; ++r) {
        int row = tileM * 16 + r + half * 8;
        int l = row >> 6, n = row & 63;
        int b_ = l >> 8, rem = l & 255;
        int gy = rem >> 4, gx = rem & 15;
        int by = n >> 3,  bx = n & 7;
        int y = gy * 8 + by, x = gx * 8 + bx;
        if (y < 126 && x < 126)
            out[(((size_t)b_ * 128 + j) * 126 + y) * 126 + x] = acc[r] + bias;
    }
}

extern "C" void kernel_launch(void* const* d_in, const int* in_sizes, int n_in,
                              void* d_out, int out_size, void* d_ws, size_t ws_size,
                              hipStream_t stream) {
    const float* x     = (const float*)d_in[0];   // (2,128,126,126)
    const float* w_in  = (const float*)d_in[1];   // (384,128)
    const float* b_in  = (const float*)d_in[2];   // (384,)
    const float* w_out = (const float*)d_in[3];   // (128,128)
    const float* b_out = (const float*)d_in[4];   // (128,)
    float* out = (float*)d_out;

    char* ws = (char*)d_ws;
    f16* Wi = (f16*)(ws);                         //    98,304 B
    f16* Wo = (f16*)(ws + 98304);                 //    32,768 B
    f16* Xw = (f16*)(ws + 131072);                // 8,388,608 B
    f16* Qp = (f16*)(ws + 8519680);               // 16,777,216 B
    f16* Kp = (f16*)(ws + 25296896);              // 16,777,216 B
    f16* Vt = (f16*)(ws + 42074112);              // 8,388,608 B
    f16* Of = (f16*)(ws + 50462720);              // 8,388,608 B  (end ~58.9 MB)

    prep_weights<<<192, 256, 0, stream>>>(w_in, w_out, Wi, Wo);
    window_kernel<<<16384, 256, 0, stream>>>(x, Xw);
    qkv_gemm<<<dim3(256, 24), 256, 0, stream>>>(Xw, Wi, b_in, Qp, Kp, Vt);
    attn_kernel<<<512, 256, 81920, stream>>>(Qp, Kp, Vt, Of);
    out_gemm<<<dim3(256, 8), 256, 0, stream>>>(Of, Wo, b_out, out);
}
